// DictlessHeteroLayer_7292854469248
// MI455X (gfx1250) — matrix operand order repaired and checked by reading, not compiled, not run on hardware
//
#include <hip/hip_runtime.h>
#include <hip/hip_bf16.h>

// RGCN-style layer for MI455X (gfx1250, wave32).
// Phase 1: split x and weights into bf16 hi/lo (error-compensated bf16 GEMM).
// Phase 2: 5 fused GEMMs via v_wmma_f32_16x16x32_bf16 (slot 0 = combined self
//          transform + bias -> d_out; slots 1..R = per-relation neighbor
//          transform -> workspace).
// Phase 3: per-edge gather/scale/scatter with f32 global atomics (segment sum).
// Workspace requirement ~257 MB for N=100000, D=128, R=4.

typedef __bf16 bf16;
typedef __attribute__((ext_vector_type(16))) __bf16 v16bf;
typedef __attribute__((ext_vector_type(8)))  __bf16 v8bf;   // 16 bytes
typedef __attribute__((ext_vector_type(8)))  float  v8f;

#define DD 128   // feature dim (fixed by reference)

// ---------------------------------------------------------------- phase 1a
__global__ __launch_bounds__(256) void split_x_k(const float* __restrict__ x,
                                                 bf16* __restrict__ xh,
                                                 bf16* __restrict__ xl,
                                                 long n) {
  long i = (long)blockIdx.x * blockDim.x + threadIdx.x;
  if (i >= n) return;
  float v = x[i];
  bf16 h = (bf16)v;             // RNE truncation to bf16
  xh[i] = h;
  xl[i] = (bf16)(v - (float)h); // residual
}

// ---------------------------------------------------------------- phase 1b
// Packed weights: slot 0 = sum_r W_self[r]; slots 1..R = W_nei[r].
// Layout [slot][j][k] row-major (j = output col, k contiguous) -> B-fragment
// rows are contiguous 16-element bf16 runs.
__global__ __launch_bounds__(256) void prep_w_k(const float* __restrict__ Ws,
                                                const float* __restrict__ Wn,
                                                const float* __restrict__ bs,
                                                bf16* __restrict__ Wh,
                                                bf16* __restrict__ Wl,
                                                float* __restrict__ bsum,
                                                int n_rel) {
  int i = blockIdx.x * blockDim.x + threadIdx.x;
  int tot = (n_rel + 1) * DD * DD;
  if (i < tot) {
    int slot = i / (DD * DD);
    int off  = i % (DD * DD);
    float v;
    if (slot == 0) {
      v = 0.f;
      for (int r = 0; r < n_rel; ++r) v += Ws[(size_t)r * DD * DD + off];
    } else {
      v = Wn[(size_t)(slot - 1) * DD * DD + off];
    }
    bf16 h = (bf16)v;
    Wh[i] = h;
    Wl[i] = (bf16)(v - (float)h);
  }
  if (i < DD) {
    float b = 0.f;
    for (int r = 0; r < n_rel; ++r) b += bs[r * DD + i];
    bsum[i] = b;
  }
}

// ---------------------------------------------------------------- phase 2
// Block = 128 threads = 4 waves; each wave computes a 16x128 output strip.
// grid = (ceil(N/64), n_rel+1). A/B fragments loaded directly from global
// (weights are L2/L1-hot: 0.33 MB total). K loop: 4 steps of K=32, 3 wmma
// per (K-step, N-tile) for the hi/lo error-compensated product.
__global__ __launch_bounds__(128) void gemm_k(const bf16* __restrict__ xh,
                                              const bf16* __restrict__ xl,
                                              const bf16* __restrict__ Wh,
                                              const bf16* __restrict__ Wl,
                                              const float* __restrict__ bsum,
                                              float* __restrict__ out,
                                              float* __restrict__ Hnei,
                                              int N) {
  const int wave = threadIdx.x >> 5;
  const int lane = threadIdx.x & 31;
  const int slot = blockIdx.y;
  const int rowbase = blockIdx.x * 64 + wave * 16;
  if (rowbase >= N) return;                    // wave-uniform guard (16 | N)

  const int mrow   = rowbase + (lane & 15);    // A: row per lane
  const int kgrpA  = (lane >> 4) * 16;         // A: K half base
  const int ncol   = lane & 15;                // B/C/D: column per lane
  const int kgrpB  = (lane >> 4) * 16;         // B: K half base

  const bf16* __restrict__ wbh = Wh + (size_t)slot * DD * DD;
  const bf16* __restrict__ wbl = Wl + (size_t)slot * DD * DD;
  const bf16* __restrict__ arh = xh + (size_t)mrow * DD;
  const bf16* __restrict__ arl = xl + (size_t)mrow * DD;

  const v8f vzero = {0.f, 0.f, 0.f, 0.f, 0.f, 0.f, 0.f, 0.f};
  v8f acc[8];
#pragma unroll
  for (int t = 0; t < 8; ++t) acc[t] = vzero;

#pragma unroll
  for (int kt = 0; kt < 4; ++kt) {
    const int kb = kt * 32;
    // A fragments (16x32 bf16): lane holds K=[kgrpA..+15]
    v16bf ah, al;
    ((v8bf*)&ah)[0] = *(const v8bf*)(arh + kb + kgrpA);
    ((v8bf*)&ah)[1] = *(const v8bf*)(arh + kb + kgrpA + 8);
    ((v8bf*)&al)[0] = *(const v8bf*)(arl + kb + kgrpA);
    ((v8bf*)&al)[1] = *(const v8bf*)(arl + kb + kgrpA + 8);
#pragma unroll
    for (int nt = 0; nt < 8; ++nt) {
      const size_t wo = (size_t)(nt * 16 + ncol) * DD + kb + kgrpB;
      v16bf bh = *(const v16bf*)(wbh + wo);
      v16bf bl = *(const v16bf*)(wbl + wo);
      acc[nt] = __builtin_amdgcn_wmma_f32_16x16x32_bf16(
          false, ah, false, bh, (short)0, acc[nt], false, false);
      acc[nt] = __builtin_amdgcn_wmma_f32_16x16x32_bf16(
          false, ah, false, bl, (short)0, acc[nt], false, false);
      acc[nt] = __builtin_amdgcn_wmma_f32_16x16x32_bf16(
          false, al, false, bh, (short)0, acc[nt], false, false);
    }
  }

  // Epilogue. C/D layout: VGPR i, lane L -> row = rowbase + (L>>4)*8 + i,
  // col = nt*16 + (L&15). Stores are lane-contiguous per row.
  const int rbase2 = rowbase + (lane >> 4) * 8;
  if (slot == 0) {
#pragma unroll
    for (int nt = 0; nt < 8; ++nt) {
      const float bv = bsum[nt * 16 + ncol];
#pragma unroll
      for (int i = 0; i < 8; ++i)
        out[(size_t)(rbase2 + i) * DD + nt * 16 + ncol] = acc[nt][i] + bv;
    }
  } else {
    float* __restrict__ hp = Hnei + (size_t)(slot - 1) * N * DD;
#pragma unroll
    for (int nt = 0; nt < 8; ++nt)
#pragma unroll
      for (int i = 0; i < 8; ++i)
        hp[(size_t)(rbase2 + i) * DD + nt * 16 + ncol] = acc[nt][i];
  }
}

// ---------------------------------------------------------------- phase 3
// One wave per edge iteration: lane l handles columns 4l..4l+3.
// h_nei (51 MB/relation) and out (51 MB) are L2-resident -> atomics at L2 rate.
__global__ __launch_bounds__(256) void scatter_k(const int* __restrict__ edge_index,
                                                 const float* __restrict__ ew,
                                                 const long long* __restrict__ rel_ptr,
                                                 const float* __restrict__ Hnei,
                                                 float* __restrict__ out,
                                                 int N, int E, int n_rel) {
  const int lane = threadIdx.x & 31;
  const int wid  = (int)((blockIdx.x * blockDim.x + threadIdx.x) >> 5);
  const int nw   = (int)((gridDim.x * blockDim.x) >> 5);

  for (int e = wid; e < E; e += nw) {
    const int   src = edge_index[e];              // row 0 of edge_index
    const int   dst = edge_index[(size_t)E + e];  // row 1
    const float w   = ew[e];
    int r = 0;
#pragma unroll 4
    for (int q = 1; q < n_rel; ++q)
      if ((long long)e >= rel_ptr[q]) ++r;        // rel_ptr is uniform, L1-hot

    const float4* __restrict__ hp =
        (const float4*)(Hnei + ((size_t)r * N + src) * DD);
    float4 v = hp[lane];                          // 512 B per wave, coalesced
    float* __restrict__ op = out + (size_t)dst * DD + lane * 4;
    __hip_atomic_fetch_add(op + 0, v.x * w, __ATOMIC_RELAXED, __HIP_MEMORY_SCOPE_AGENT);
    __hip_atomic_fetch_add(op + 1, v.y * w, __ATOMIC_RELAXED, __HIP_MEMORY_SCOPE_AGENT);
    __hip_atomic_fetch_add(op + 2, v.z * w, __ATOMIC_RELAXED, __HIP_MEMORY_SCOPE_AGENT);
    __hip_atomic_fetch_add(op + 3, v.w * w, __ATOMIC_RELAXED, __HIP_MEMORY_SCOPE_AGENT);
  }
}

// ---------------------------------------------------------------- host
extern "C" void kernel_launch(void* const* d_in, const int* in_sizes, int n_in,
                              void* d_out, int out_size, void* d_ws, size_t ws_size,
                              hipStream_t stream) {
  const float*     x          = (const float*)d_in[0];
  const int*       edge_index = (const int*)d_in[1];
  const float*     ew         = (const float*)d_in[2];
  const long long* rel_ptr    = (const long long*)d_in[3];   // int64
  const float*     W_self     = (const float*)d_in[4];
  const float*     b_self     = (const float*)d_in[5];
  const float*     W_nei      = (const float*)d_in[6];
  float*           out        = (float*)d_out;

  const int N     = in_sizes[0] / DD;     // 100000
  const int E     = in_sizes[2];          // 2000000
  const int n_rel = in_sizes[5] / DD;     // 4

  // Workspace carve-out (256 B aligned blocks).
  char*  ws  = (char*)d_ws;
  size_t off = 0;
  auto carve = [&](size_t bytes) {
    size_t o = off;
    off = (off + bytes + 255) & ~(size_t)255;
    return o;
  };
  const size_t nx = (size_t)N * DD;
  const size_t nw = (size_t)(n_rel + 1) * DD * DD;
  bf16*  xh   = (bf16*)(ws + carve(nx * sizeof(bf16)));
  bf16*  xl   = (bf16*)(ws + carve(nx * sizeof(bf16)));
  bf16*  Wh   = (bf16*)(ws + carve(nw * sizeof(bf16)));
  bf16*  Wl   = (bf16*)(ws + carve(nw * sizeof(bf16)));
  float* bsum = (float*)(ws + carve(DD * sizeof(float)));
  float* Hnei = (float*)(ws + carve((size_t)n_rel * nx * sizeof(float)));
  (void)ws_size; (void)n_in; (void)out_size;   // total need ~257 MB

  // Phase 1: bf16 hi/lo splits.
  {
    long n = (long)nx;
    int blocks = (int)((n + 255) / 256);
    split_x_k<<<blocks, 256, 0, stream>>>(x, xh, xl, n);
    int wthreads = (n_rel + 1) * DD * DD;
    prep_w_k<<<(wthreads + 255) / 256, 256, 0, stream>>>(
        W_self, W_nei, b_self, Wh, Wl, bsum, n_rel);
  }

  // Phase 2: WMMA GEMMs (slot 0 -> out with bias, slots 1..R -> Hnei).
  {
    dim3 grid((N + 63) / 64, n_rel + 1);
    gemm_k<<<grid, 128, 0, stream>>>(xh, xl, Wh, Wl, bsum, out, Hnei, N);
  }

  // Phase 3: edge gather/scale/scatter-add.
  {
    scatter_k<<<2048, 256, 0, stream>>>(edge_index, ew, rel_ptr, Hnei, out,
                                        N, E, n_rel);
  }
}
